// BiMultiHeadAttention_1477468750338
// MI455X (gfx1250) — matrix operand order repaired and checked by reading, not compiled, not run on hardware
//
#include <hip/hip_runtime.h>

// ---------------- problem constants ----------------
#define BSZ   4
#define TI    16384
#define TL    256
#define EMB   1024
#define NHEAD 8
#define HD    128
#define QK_SCALE 0.08838834764831845f   // 128^-0.5
#define NEGF (-3.0e38f)

typedef unsigned short u16;
typedef __bf16 bf16x16 __attribute__((ext_vector_type(16)));
typedef float  f32x8   __attribute__((ext_vector_type(8)));
typedef u16    u16x8   __attribute__((ext_vector_type(8)));

union FragU { bf16x16 bf; u16x8 h[2]; };

static __device__ __forceinline__ u16 f2bf(float f) {
  unsigned u = __float_as_uint(f);
  u += 0x7fffu + ((u >> 16) & 1u);          // round-to-nearest-even
  return (u16)(u >> 16);
}

static __device__ __forceinline__ f32x8 wmma_bf16(bf16x16 a, bf16x16 b, f32x8 c) {
  return __builtin_amdgcn_wmma_f32_16x16x32_bf16(false, a, false, b, (short)0, c,
                                                 false, false);
}

// ---- CDNA5 async global->LDS copy (ASYNCcnt path, bypasses VGPRs) ----------
// VDST operand is the LDS *byte offset* relative to the wave's LDS base. All
// kernels using this keep their entire LDS in the dynamic extern-shared region,
// which starts at offset 0 — so we pass pure integer offsets (no LDS symbol in
// the asm operand, avoiding unsupported addrspacecast relocations).
static __device__ __forceinline__ void async_g2l_b128(const void* gsrc,
                                                      unsigned lds_byte_off) {
  asm volatile("global_load_async_to_lds_b128 %0, %1, off"
               :
               : "v"(lds_byte_off),
                 "v"((unsigned long long)(size_t)gsrc)
               : "memory");
}
static __device__ __forceinline__ void wait_async0() {
  asm volatile("s_wait_asynccnt 0" ::: "memory");
}

// A fragment: 16x32 bf16 tile from LDS, row-major [M][K], stride lda (halves).
// lane<16: M=lane, K groups {0-7,16-23}; lane>=16: K groups {8-15,24-31}.
static __device__ __forceinline__ bf16x16 ldsA(const u16* base, int lda, int m0, int k0) {
  int lane = threadIdx.x & 31;
  const u16* p = base + (m0 + (lane & 15)) * lda + k0 + ((lane >> 4) << 3);
  FragU f;
  f.h[0] = *(const u16x8*)(p);
  f.h[1] = *(const u16x8*)(p + 16);
  return f.bf;
}

// B fragment: 32x16 bf16 from an [N][K] row-major LDS tile, stride ldb (halves).
// lane<16: N=lane, K groups {0-7,16-23}; lane>=16: K groups {8-15,24-31}.
static __device__ __forceinline__ bf16x16 ldsB(const u16* base, int ldb, int n0, int k0) {
  int lane = threadIdx.x & 31;
  const u16* p = base + (n0 + (lane & 15)) * ldb + k0 + ((lane >> 4) << 3);
  FragU f;
  f.h[0] = *(const u16x8*)(p);
  f.h[1] = *(const u16x8*)(p + 16);
  return f.bf;
}

// ---------------- fp32 -> bf16 conversion ----------------
__global__ void f32_to_bf16_kernel(const float* __restrict__ src,
                                   u16* __restrict__ dst, int n) {
  for (int i = blockIdx.x * blockDim.x + threadIdx.x; i < n;
       i += gridDim.x * blockDim.x)
    dst[i] = f2bf(src[i]);
}

// ---------------- generic GEMM: C[M,N] = A[M,K] @ W[N,K]^T, +bias, *scale ----
// A, W bf16; output bf16 (Cbf) or f32 (Cf). Block tile 128x64, 8 waves of 32x32.
// Double-buffered dynamic-LDS tiles filled by async global->LDS DMA.
// LDS layout (bytes): As0 @0 (10240), As1 @10240, Ws0 @20480 (5120), Ws1 @25600.
#define GE_AS0 0u
#define GE_AS1 10240u
#define GE_WS0 20480u
#define GE_WS1 25600u
#define GE_SMEM 30720

__global__ __launch_bounds__(256) void gemm_bias_kernel(
    const u16* __restrict__ A, const u16* __restrict__ W,
    const float* __restrict__ bias, u16* __restrict__ Cbf, float* __restrict__ Cf,
    int M, int N, int K, float scale) {
  extern __shared__ char gsm[];
  const int tid = threadIdx.x;
  const int lane = tid & 31, w = tid >> 5;
  const int wm = w & 3, wn = w >> 2;
  const int m0 = blockIdx.y * 128, n0 = blockIdx.x * 64;

  auto stage = [&](int buf, int k0) {
    const unsigned aBase = buf ? GE_AS1 : GE_AS0;
    const unsigned wBase = buf ? GE_WS1 : GE_WS0;
#pragma unroll
    for (int it = 0; it < 2; ++it) {           // A tile 128x32
      int idx = tid + (it << 8);
      int r = idx >> 2, c8 = (idx & 3) << 3;
      async_g2l_b128(A + (size_t)(m0 + r) * K + k0 + c8,
                     aBase + (unsigned)(r * 40 + c8) * 2u);
    }
    {                                          // W tile 64x32 ([N][K])
      int r = tid >> 2, c8 = (tid & 3) << 3;
      async_g2l_b128(W + (size_t)(n0 + r) * K + k0 + c8,
                     wBase + (unsigned)(r * 40 + c8) * 2u);
    }
  };

  f32x8 acc[2][2] = {};
  stage(0, 0);
  wait_async0();
  __syncthreads();
  int cur = 0;
  for (int k0 = 0; k0 < K; k0 += 32) {
    if (k0 + 32 < K) stage(cur ^ 1, k0 + 32);  // overlap DMA with WMMA
    const u16* Asc = (const u16*)(gsm + (cur ? GE_AS1 : GE_AS0));
    const u16* Wsc = (const u16*)(gsm + (cur ? GE_WS1 : GE_WS0));
    bf16x16 a0 = ldsA(Asc, 40, wm * 32, 0);
    bf16x16 a1 = ldsA(Asc, 40, wm * 32 + 16, 0);
    bf16x16 b0 = ldsB(Wsc, 40, wn * 32, 0);
    bf16x16 b1 = ldsB(Wsc, 40, wn * 32 + 16, 0);
    acc[0][0] = wmma_bf16(a0, b0, acc[0][0]);
    acc[0][1] = wmma_bf16(a0, b1, acc[0][1]);
    acc[1][0] = wmma_bf16(a1, b0, acc[1][0]);
    acc[1][1] = wmma_bf16(a1, b1, acc[1][1]);
    wait_async0();
    __syncthreads();
    cur ^= 1;
  }
  const int ln = lane & 15, hi = lane >> 4;
#pragma unroll
  for (int i = 0; i < 2; ++i)
#pragma unroll
    for (int j = 0; j < 2; ++j) {
      int n = n0 + wn * 32 + j * 16 + ln;
      float bv = bias[n];
#pragma unroll
      for (int r = 0; r < 8; ++r) {
        int m = m0 + wm * 32 + i * 16 + r + hi * 8;
        float v = (acc[i][j][r] + bv) * scale;   // reference: (x@W^T + b) * scale
        if (Cbf) Cbf[(size_t)m * N + n] = f2bf(v);
        else     Cf [(size_t)m * N + n] = v;
      }
    }
}

// ---------------- direction 1: out_v head attention (keys = 256, fused) ------
// LDS byte offsets (dynamic smem, 254976 B total, < 320KB/WGP)
#define AV_QS 0        // 64x136 u16
#define AV_KS 17408    // 256x136 u16  ([j][d] row-major -> B of Q@K^T)
#define AV_VT 87040    // 128x264 u16  (V transposed [d][j] -> B of P@V)
#define AV_S  154624   // 64x260 f32
#define AV_P  221184   // 64x264 u16
#define AV_SMEM 254976

__global__ __launch_bounds__(256) void attn_v_kernel(
    const u16* __restrict__ Qm, const u16* __restrict__ Km,
    const u16* __restrict__ Vm, const unsigned char* __restrict__ maskL,
    u16* __restrict__ O) {
  extern __shared__ char smem[];
  u16*   Qs = (u16*)(smem + AV_QS);
  u16*   Ks = (u16*)(smem + AV_KS);
  u16*   Vt = (u16*)(smem + AV_VT);
  float* S  = (float*)(smem + AV_S);
  u16*   P  = (u16*)(smem + AV_P);
  const int tid = threadIdx.x, lane = tid & 31, w = tid >> 5;
  const int bh = blockIdx.y, b = bh >> 3, h = bh & 7;
  const int i0 = blockIdx.x * 64;
  const size_t qoff = ((size_t)b * TI) * EMB + h * HD;
  const size_t koff = ((size_t)b * TL) * EMB + h * HD;

#pragma unroll
  for (int it = 0; it < 4; ++it) {             // stage Q 64x128 (async DMA)
    int idx = tid + (it << 8);
    int r = idx >> 4, c8 = (idx & 15) << 3;
    async_g2l_b128(Qm + qoff + (size_t)(i0 + r) * EMB + c8,
                   AV_QS + (unsigned)(r * 136 + c8) * 2u);
  }
  {                                            // stage K (async) and V (transpose)
    int j = tid;
#pragma unroll
    for (int d0 = 0; d0 < HD; d0 += 8) {
      async_g2l_b128(Km + koff + (size_t)j * EMB + d0,
                     AV_KS + (unsigned)(j * 136 + d0) * 2u);
      u16x8 vv = *(const u16x8*)(Vm + koff + (size_t)j * EMB + d0);
#pragma unroll
      for (int e = 0; e < 8; ++e) Vt[(d0 + e) * 264 + j] = vv[e];
    }
  }
  wait_async0();
  __syncthreads();

  {  // scores S[64][256] = Q @ K^T  (scale folded into q already)
    const int mt = w & 3, ntb = (w >> 2) * 8;
    bf16x16 af[4];
#pragma unroll
    for (int ks = 0; ks < 4; ++ks) af[ks] = ldsA(Qs, 136, mt * 16, ks * 32);
    const int ln = lane & 15, hi = lane >> 4;
    for (int t = 0; t < 8; ++t) {
      int nt = ntb + t;
      f32x8 c = {};
#pragma unroll
      for (int ks = 0; ks < 4; ++ks)
        c = wmma_bf16(af[ks], ldsB(Ks, 136, nt * 16, ks * 32), c);
#pragma unroll
      for (int r = 0; r < 8; ++r)
        S[(mt * 16 + r + hi * 8) * 260 + nt * 16 + ln] = c[r];
    }
  }
  __syncthreads();

  // per-row stable softmax over 256 keys (global-max+clip of the reference is a
  // shift-invariance no-op here); masked keys -> 0 probability
#pragma unroll
  for (int rr = 0; rr < 8; ++rr) {
    int r = w * 8 + rr;
    float vals[8]; float m = NEGF;
#pragma unroll
    for (int t = 0; t < 8; ++t) {
      int j = lane + t * 32;
      float s = S[r * 260 + j];
      if (maskL[b * TL + j]) s = NEGF;
      vals[t] = s; m = fmaxf(m, s);
    }
#pragma unroll
    for (int off = 16; off >= 1; off >>= 1) m = fmaxf(m, __shfl_xor(m, off, 32));
    float sum = 0.f; float e[8];
#pragma unroll
    for (int t = 0; t < 8; ++t) {
      e[t] = (vals[t] <= NEGF) ? 0.f : __expf(vals[t] - m);
      sum += e[t];
    }
#pragma unroll
    for (int off = 16; off >= 1; off >>= 1) sum += __shfl_xor(sum, off, 32);
    float inv = sum > 0.f ? 1.f / sum : 0.f;
#pragma unroll
    for (int t = 0; t < 8; ++t) P[r * 264 + lane + t * 32] = f2bf(e[t] * inv);
  }
  __syncthreads();

  {  // O[64][128] = P @ V
    const int mt = w & 3, ntb = (w >> 2) * 4;
    f32x8 acc[4] = {};
    for (int ks = 0; ks < 8; ++ks) {
      bf16x16 a = ldsA(P, 264, mt * 16, ks * 32);
#pragma unroll
      for (int t = 0; t < 4; ++t)
        acc[t] = wmma_bf16(a, ldsB(Vt, 264, (ntb + t) * 16, ks * 32), acc[t]);
    }
    const int ln = lane & 15, hi = lane >> 4;
#pragma unroll
    for (int t = 0; t < 4; ++t)
#pragma unroll
      for (int r = 0; r < 8; ++r) {
        int m = i0 + mt * 16 + r + hi * 8;
        int n = (ntb + t) * 16 + ln;
        O[qoff + (size_t)m * EMB + n] = f2bf(acc[t][r]);
      }
  }
}

// ---------------- direction 2 pass A: row max/sum over the 16384 axis --------
// Qc chunk is double-buffered and filled by async DMA while WMMA runs.
#define ST_Q2  0        // 64x136 u16
#define ST_QC0 17408    // 256x136 u16 chunk buffer 0
#define ST_QC1 87040    // 256x136 u16 chunk buffer 1
#define ST_S   156672   // 64x260 f32
#define ST_SMEM 223232

__global__ __launch_bounds__(256) void attn_l_stats_kernel(
    const u16* __restrict__ Km /*queries*/, const u16* __restrict__ Qm /*keys*/,
    const unsigned char* __restrict__ maskV,
    float* __restrict__ SM, float* __restrict__ SS) {
  extern __shared__ char smem[];
  u16*   Q2 = (u16*)(smem + ST_Q2);
  float* S  = (float*)(smem + ST_S);
  const int tid = threadIdx.x, lane = tid & 31, w = tid >> 5;
  const int bh = blockIdx.y, b = bh >> 3, h = bh & 7;
  const int r0 = blockIdx.x * 64;
  const size_t koff = ((size_t)b * TL) * EMB + h * HD;
  const size_t qoff = ((size_t)b * TI) * EMB + h * HD;

#pragma unroll
  for (int it = 0; it < 4; ++it) {
    int idx = tid + (it << 8);
    int r = idx >> 4, c8 = (idx & 15) << 3;
    async_g2l_b128(Km + koff + (size_t)(r0 + r) * EMB + c8,
                   ST_Q2 + (unsigned)(r * 136 + c8) * 2u);
  }
  {                                            // prime chunk 0
    int j = tid;
#pragma unroll
    for (int d0 = 0; d0 < HD; d0 += 8)
      async_g2l_b128(Qm + qoff + (size_t)j * EMB + d0,
                     ST_QC0 + (unsigned)(j * 136 + d0) * 2u);
  }
  wait_async0();
  __syncthreads();

  const int mt = w & 3, ntb = (w >> 2) * 8;
  bf16x16 af[4];
#pragma unroll
  for (int ks = 0; ks < 4; ++ks) af[ks] = ldsA(Q2, 136, mt * 16, ks * 32);

  float rm[8], rs[8];
#pragma unroll
  for (int rr = 0; rr < 8; ++rr) { rm[rr] = NEGF; rs[rr] = 0.f; }

  int cur = 0;
  for (int ic = 0; ic < TI; ic += 256) {
    if (ic + 256 < TI) {                       // prefetch next chunk via DMA
      const unsigned qcb = (cur ^ 1) ? ST_QC1 : ST_QC0;
      int j = tid;
#pragma unroll
      for (int d0 = 0; d0 < HD; d0 += 8)
        async_g2l_b128(Qm + qoff + (size_t)(ic + 256 + j) * EMB + d0,
                       qcb + (unsigned)(j * 136 + d0) * 2u);
    }
    {
      const u16* Qcur = (const u16*)(smem + (cur ? ST_QC1 : ST_QC0));
      const int ln = lane & 15, hi = lane >> 4;
      for (int t = 0; t < 8; ++t) {
        int nt = ntb + t;
        f32x8 c = {};
#pragma unroll
        for (int ks = 0; ks < 4; ++ks)
          c = wmma_bf16(af[ks], ldsB(Qcur, 136, nt * 16, ks * 32), c);
#pragma unroll
        for (int r = 0; r < 8; ++r)
          S[(mt * 16 + r + hi * 8) * 260 + nt * 16 + ln] = c[r];
      }
    }
    __syncthreads();
#pragma unroll
    for (int rr = 0; rr < 8; ++rr) {  // online (m, sum) merge, scalar per row
      int r = w * 8 + rr;
      float vals[8]; float cm = NEGF;
#pragma unroll
      for (int t = 0; t < 8; ++t) {
        int j = lane + t * 32;
        float s = S[r * 260 + j];
        if (maskV[(size_t)b * TI + ic + j]) s = NEGF;
        vals[t] = s; cm = fmaxf(cm, s);
      }
#pragma unroll
      for (int off = 16; off >= 1; off >>= 1) cm = fmaxf(cm, __shfl_xor(cm, off, 32));
      float mn = fmaxf(rm[rr], cm);
      float cs = 0.f;
#pragma unroll
      for (int t = 0; t < 8; ++t)
        cs += (vals[t] <= NEGF) ? 0.f : __expf(vals[t] - mn);
#pragma unroll
      for (int off = 16; off >= 1; off >>= 1) cs += __shfl_xor(cs, off, 32);
      rs[rr] = rs[rr] * __expf(rm[rr] - mn) + cs;
      rm[rr] = mn;
    }
    wait_async0();
    __syncthreads();
    cur ^= 1;
  }
  if (lane == 0) {
#pragma unroll
    for (int rr = 0; rr < 8; ++rr) {
      int r = bh * TL + r0 + w * 8 + rr;
      SM[r] = rm[rr]; SS[r] = rs[rr];
    }
  }
}

// ---------------- direction 2 pass B: out_l = softmax(S2) @ val_v ------------
#define AL_Q2 0
#define AL_QC 17408
#define AL_VT 87040    // 128x264 u16 (val_v chunk transposed)
#define AL_S  154624
#define AL_P  221184
#define AL_SM 254976
#define AL_SI 255232
#define AL_SMEM 255488

__global__ __launch_bounds__(256) void attn_l_out_kernel(
    const u16* __restrict__ Km, const u16* __restrict__ Qm,
    const u16* __restrict__ Vm, const unsigned char* __restrict__ maskV,
    const float* __restrict__ SM, const float* __restrict__ SS,
    u16* __restrict__ O) {
  extern __shared__ char smem[];
  u16*   Q2 = (u16*)(smem + AL_Q2);
  u16*   Qc = (u16*)(smem + AL_QC);
  u16*   Vt = (u16*)(smem + AL_VT);
  float* S  = (float*)(smem + AL_S);
  u16*   P  = (u16*)(smem + AL_P);
  float* sm = (float*)(smem + AL_SM);
  float* si = (float*)(smem + AL_SI);
  const int tid = threadIdx.x, lane = tid & 31, w = tid >> 5;
  const int bh = blockIdx.y, b = bh >> 3, h = bh & 7;
  const int r0 = blockIdx.x * 64;
  const size_t koff = ((size_t)b * TL) * EMB + h * HD;
  const size_t qoff = ((size_t)b * TI) * EMB + h * HD;

#pragma unroll
  for (int it = 0; it < 4; ++it) {
    int idx = tid + (it << 8);
    int r = idx >> 4, c8 = (idx & 15) << 3;
    async_g2l_b128(Km + koff + (size_t)(r0 + r) * EMB + c8,
                   AL_Q2 + (unsigned)(r * 136 + c8) * 2u);
  }
  if (tid < 64) {
    float s = SS[bh * TL + r0 + tid];
    sm[tid] = SM[bh * TL + r0 + tid];
    si[tid] = s > 0.f ? 1.f / s : 0.f;
  }
  wait_async0();
  __syncthreads();

  const int mt = w & 3, sntb = (w >> 2) * 8, pntb = (w >> 2) * 4;
  bf16x16 af[4];
#pragma unroll
  for (int ks = 0; ks < 4; ++ks) af[ks] = ldsA(Q2, 136, mt * 16, ks * 32);

  f32x8 acc[4] = {};  // persistent across key chunks (fixed m,sum -> no rescale)

  for (int ic = 0; ic < TI; ic += 256) {
    __syncthreads();
    {
      int j = tid;
#pragma unroll
      for (int d0 = 0; d0 < HD; d0 += 8)   // async copy of q chunk
        async_g2l_b128(Qm + qoff + (size_t)(ic + j) * EMB + d0,
                       AL_QC + (unsigned)(j * 136 + d0) * 2u);
#pragma unroll
      for (int d0 = 0; d0 < HD; d0 += 8) { // manual transposed copy of val_v
        u16x8 vv = *(const u16x8*)(Vm + qoff + (size_t)(ic + j) * EMB + d0);
#pragma unroll
        for (int e = 0; e < 8; ++e) Vt[(d0 + e) * 264 + j] = vv[e];
      }
    }
    wait_async0();
    __syncthreads();
    {
      const int ln = lane & 15, hi = lane >> 4;
      for (int t = 0; t < 8; ++t) {
        int nt = sntb + t;
        f32x8 c = {};
#pragma unroll
        for (int ks = 0; ks < 4; ++ks)
          c = wmma_bf16(af[ks], ldsB(Qc, 136, nt * 16, ks * 32), c);
#pragma unroll
        for (int r = 0; r < 8; ++r)
          S[(mt * 16 + r + hi * 8) * 260 + nt * 16 + ln] = c[r];
      }
    }
    __syncthreads();
    for (int it = 0; it < 64; ++it) {   // P = exp(S - m) / sum, masked -> 0
      int idx = tid + (it << 8);
      int r = idx >> 8, j = idx & 255;
      float s = S[r * 260 + j];
      float e = maskV[(size_t)b * TI + ic + j] ? 0.f : __expf(s - sm[r]) * si[r];
      P[r * 264 + j] = f2bf(e);
    }
    __syncthreads();
    for (int ks = 0; ks < 8; ++ks) {
      bf16x16 a = ldsA(P, 264, mt * 16, ks * 32);
#pragma unroll
      for (int t = 0; t < 4; ++t)
        acc[t] = wmma_bf16(a, ldsB(Vt, 264, (pntb + t) * 16, ks * 32), acc[t]);
    }
  }
  const int ln = lane & 15, hi = lane >> 4;
#pragma unroll
  for (int t = 0; t < 4; ++t)
#pragma unroll
    for (int r = 0; r < 8; ++r) {
      int m = r0 + mt * 16 + r + hi * 8;
      int n = (pntb + t) * 16 + ln;
      O[koff + (size_t)m * EMB + n] = f2bf(acc[t][r]);
    }
}

// ---------------- workspace layout (bytes; needs ~426 MB of d_ws) ------------
#define WS_VBF   0u
#define WS_LBF   33554432u
#define WS_WBF   34078720u   // 6 weights x 524288 B each
#define WS_QBF   37224448u
#define WS_KBF   171442176u
#define WS_VVBF  173539328u
#define WS_VLBF  307757056u
#define WS_OVBF  309854208u
#define WS_OLBF  444071936u
#define WS_SM    446169088u
#define WS_SS    446201856u

extern "C" void kernel_launch(void* const* d_in, const int* in_sizes, int n_in,
                              void* d_out, int out_size, void* d_ws, size_t ws_size,
                              hipStream_t stream) {
  (void)in_sizes; (void)n_in; (void)out_size; (void)ws_size;
  const float* v_in = (const float*)d_in[0];
  const float* l_in = (const float*)d_in[1];
  const unsigned char* mask_v = (const unsigned char*)d_in[2];  // jnp bool = 1B
  const unsigned char* mask_l = (const unsigned char*)d_in[3];
  const float* vpw = (const float*)d_in[4];  const float* vpb = (const float*)d_in[5];
  const float* lpw = (const float*)d_in[6];  const float* lpb = (const float*)d_in[7];
  const float* vvw = (const float*)d_in[8];  const float* vvb = (const float*)d_in[9];
  const float* vlw = (const float*)d_in[10]; const float* vlb = (const float*)d_in[11];
  const float* ovw = (const float*)d_in[12]; const float* ovb = (const float*)d_in[13];
  const float* olw = (const float*)d_in[14]; const float* olb = (const float*)d_in[15];

  char* ws = (char*)d_ws;
  u16* v_bf  = (u16*)(ws + WS_VBF);
  u16* l_bf  = (u16*)(ws + WS_LBF);
  u16* vpw_b = (u16*)(ws + WS_WBF + 0u * 524288u);
  u16* lpw_b = (u16*)(ws + WS_WBF + 1u * 524288u);
  u16* vvw_b = (u16*)(ws + WS_WBF + 2u * 524288u);
  u16* vlw_b = (u16*)(ws + WS_WBF + 3u * 524288u);
  u16* ovw_b = (u16*)(ws + WS_WBF + 4u * 524288u);
  u16* olw_b = (u16*)(ws + WS_WBF + 5u * 524288u);
  u16* q_bf  = (u16*)(ws + WS_QBF);
  u16* k_bf  = (u16*)(ws + WS_KBF);
  u16* vv_bf = (u16*)(ws + WS_VVBF);
  u16* vl_bf = (u16*)(ws + WS_VLBF);
  u16* ov_bf = (u16*)(ws + WS_OVBF);
  u16* ol_bf = (u16*)(ws + WS_OLBF);
  float* SM  = (float*)(ws + WS_SM);
  float* SS  = (float*)(ws + WS_SS);

  (void)hipFuncSetAttribute((const void*)attn_v_kernel,
                            hipFuncAttributeMaxDynamicSharedMemorySize, AV_SMEM);
  (void)hipFuncSetAttribute((const void*)attn_l_stats_kernel,
                            hipFuncAttributeMaxDynamicSharedMemorySize, ST_SMEM);
  (void)hipFuncSetAttribute((const void*)attn_l_out_kernel,
                            hipFuncAttributeMaxDynamicSharedMemorySize, AL_SMEM);

  dim3 blk(256);
  // fp32 -> bf16 staging
  f32_to_bf16_kernel<<<2048, blk, 0, stream>>>(v_in, v_bf, BSZ * TI * 256);
  f32_to_bf16_kernel<<<64,   blk, 0, stream>>>(l_in, l_bf, BSZ * TL * 256);
  f32_to_bf16_kernel<<<64,   blk, 0, stream>>>(vpw, vpw_b, EMB * 256);
  f32_to_bf16_kernel<<<64,   blk, 0, stream>>>(lpw, lpw_b, EMB * 256);
  f32_to_bf16_kernel<<<64,   blk, 0, stream>>>(vvw, vvw_b, EMB * 256);
  f32_to_bf16_kernel<<<64,   blk, 0, stream>>>(vlw, vlw_b, EMB * 256);
  f32_to_bf16_kernel<<<64,   blk, 0, stream>>>(ovw, ovw_b, 256 * EMB);
  f32_to_bf16_kernel<<<64,   blk, 0, stream>>>(olw, olw_b, 256 * EMB);

  // projections: q (scaled), k, val_v, val_l
  gemm_bias_kernel<<<dim3(EMB / 64, (BSZ * TI) / 128), blk, GE_SMEM, stream>>>(
      v_bf, vpw_b, vpb, q_bf, nullptr, BSZ * TI, EMB, 256, QK_SCALE);
  gemm_bias_kernel<<<dim3(EMB / 64, (BSZ * TL) / 128), blk, GE_SMEM, stream>>>(
      l_bf, lpw_b, lpb, k_bf, nullptr, BSZ * TL, EMB, 256, 1.0f);
  gemm_bias_kernel<<<dim3(EMB / 64, (BSZ * TI) / 128), blk, GE_SMEM, stream>>>(
      v_bf, vvw_b, vvb, vv_bf, nullptr, BSZ * TI, EMB, 256, 1.0f);
  gemm_bias_kernel<<<dim3(EMB / 64, (BSZ * TL) / 128), blk, GE_SMEM, stream>>>(
      l_bf, vlw_b, vlb, vl_bf, nullptr, BSZ * TL, EMB, 256, 1.0f);

  // fused attention, both directions
  attn_v_kernel<<<dim3(TI / 64, BSZ * NHEAD), blk, AV_SMEM, stream>>>(
      q_bf, k_bf, vl_bf, mask_l, ov_bf);
  attn_l_stats_kernel<<<dim3(TL / 64, BSZ * NHEAD), blk, ST_SMEM, stream>>>(
      k_bf, q_bf, mask_v, SM, SS);
  attn_l_out_kernel<<<dim3(TL / 64, BSZ * NHEAD), blk, AL_SMEM, stream>>>(
      k_bf, q_bf, vv_bf, mask_v, SM, SS, ol_bf);

  // output projections (f32 results straight into d_out)
  float* out_v = (float*)d_out;
  float* out_l = (float*)d_out + (size_t)BSZ * TI * 256;
  gemm_bias_kernel<<<dim3(256 / 64, (BSZ * TI) / 128), blk, GE_SMEM, stream>>>(
      ov_bf, ovw_b, ovb, nullptr, out_v, BSZ * TI, 256, EMB, 1.0f);
  gemm_bias_kernel<<<dim3(256 / 64, (BSZ * TL) / 128), blk, GE_SMEM, stream>>>(
      ol_bf, olw_b, olb, nullptr, out_l, BSZ * TL, 256, EMB, 1.0f);
}
